// Model_35399120453698
// MI455X (gfx1250) — matrix operand order repaired and checked by reading, not compiled, not run on hardware
//
#include <hip/hip_runtime.h>
#include <hip/hip_bf16.h>
#include <math.h>

typedef __attribute__((ext_vector_type(16))) _Float16 v16h;
typedef __attribute__((ext_vector_type(8)))  _Float16 v8h;
typedef __attribute__((ext_vector_type(8)))  float    v8f;

#define HD_D        10000
#define HD_B        4
#define HD_T        128
#define HD_C        32
#define HD_LEVELS   21
#define HD_KPAD     32
#define HD_M        (HD_B * HD_T)      /* 512 rows of the counts GEMM */
#define HD_NGRAM    4
#define HD_TP       (HD_T - HD_NGRAM + 1)  /* 125 */
#define HD_CLASSES  10

/* ---------------------------------------------------------------------------
 * Phase 1: per-(b,t) level histogram -> f16 counts [512][32] (K padded to 32)
 * One wave per row; LDS histogram over the 32 channels.
 * ------------------------------------------------------------------------- */
__global__ __launch_bounds__(128) void counts_kernel(
    const float* __restrict__ x,          /* [B*T*C] = [512*32] */
    _Float16* __restrict__ counts)        /* [512][32] f16 */
{
    __shared__ int hist[4][32];
    const int tid  = threadIdx.x;
    const int wave = tid >> 5;
    const int lane = tid & 31;

    hist[wave][lane] = 0;
    __syncthreads();

    const int row = blockIdx.x * 4 + wave;          /* 0..511, exact cover */
    const float v = x[row * HD_C + lane];
    int q = (int)rintf(v);                          /* idx = clip(round(x),0,20) */
    q = q < 0 ? 0 : (q > (HD_LEVELS - 1) ? (HD_LEVELS - 1) : q);
    atomicAdd(&hist[wave][q], 1);
    __syncthreads();

    _Float16 out = (_Float16)0.0f;
    if (lane < HD_LEVELS) out = (_Float16)(float)hist[wave][lane];
    counts[row * HD_KPAD + lane] = out;
}

/* ---------------------------------------------------------------------------
 * Phase 2a: signals_w f32 [21][D] -> f16 panel [32][D], rows 21..31 zero
 * ------------------------------------------------------------------------- */
__global__ __launch_bounds__(256) void sigcvt_kernel(
    const float* __restrict__ sig_f32,    /* [21][D] */
    _Float16* __restrict__ sig_f16)       /* [32][D] */
{
    const int i = blockIdx.x * 256 + threadIdx.x;
    if (i >= HD_KPAD * HD_D) return;
    const int k = i / HD_D;
    sig_f16[i] = (k < HD_LEVELS) ? (_Float16)sig_f32[i] : (_Float16)0.0f;
}

/* ---------------------------------------------------------------------------
 * Phase 2b: samples[m,n] = ts[m%T, n] * (counts @ sig_f16)[m,n]
 * WMMA f32_16x16x32_f16. Block = 8 waves = 1 M-tile x 8 N-tiles (128 cols).
 * B panel staged in LDS transposed: lds_sig[col][k] so each lane reads its
 * 16 contiguous K halves with ds_load_b128 pairs.
 * ------------------------------------------------------------------------- */
__global__ __launch_bounds__(256) void wmma_gemm_kernel(
    const _Float16* __restrict__ counts,  /* [512][32] */
    const _Float16* __restrict__ sig,     /* [32][D]   */
    const float* __restrict__ ts,         /* [T][D]    */
    float* __restrict__ samples)          /* [512][D]  */
{
    __shared__ _Float16 lds_sig[128 * HD_KPAD];   /* [col 0..127][k 0..31], 8 KB */

    const int m0  = blockIdx.x * 16;              /* 32 M-tiles  */
    const int n0  = blockIdx.y * 128;             /* 79 N-blocks */
    const int tid = threadIdx.x;

    /* Stage B tile: global row-major (contiguous in n) -> LDS col-major */
    for (int i = tid; i < 128 * HD_KPAD; i += 256) {
        const int k = i >> 7;                     /* 0..31  */
        const int c = i & 127;                    /* 0..127 */
        const int n = n0 + c;
        _Float16 v = (_Float16)0.0f;
        if (n < HD_D) v = sig[(size_t)k * HD_D + n];
        lds_sig[c * HD_KPAD + k] = v;
    }
    __syncthreads();

    const int wave  = tid >> 5;                   /* 0..7 -> N sub-tile */
    const int lane  = tid & 31;
    const int lhalf = lane >> 4;                  /* 0 or 1 */
    const int l15   = lane & 15;

    /* ---- A fragment (16-bit A 16x32, contiguous K order matching B) ----
       lane<16 : M=lane,    halves e=0..15 -> K=0..15
       lane>=16: M=lane-16, halves e=0..15 -> K=16..31   */
    const _Float16* arow = counts + (size_t)(m0 + l15) * HD_KPAD;
    const int kb = lhalf * 16;
    const v8h alo = *(const v8h*)(arow + kb);
    const v8h ahi = *(const v8h*)(arow + kb + 8);
    v16h a;
#pragma unroll
    for (int e = 0; e < 8; ++e) { a[e] = alo[e]; a[e + 8] = ahi[e]; }

    /* ---- B fragment (16-bit B 32x16 layout) ----
       lane<16 : N = n0 + wave*16 + lane,    K = 0..15  (contiguous halves)
       lane>=16: N = n0 + wave*16 + lane-16, K = 16..31                      */
    const int col_local = wave * 16 + l15;
    const int koff = lhalf * 16;
    const v16h b = *(const v16h*)(lds_sig + col_local * HD_KPAD + koff);

    v8f acc = {};
    acc = __builtin_amdgcn_wmma_f32_16x16x32_f16(
        /*neg_a=*/false, a, /*neg_b=*/false, b,
        /*c_mod=*/(short)0, acc, /*reuse_a=*/false, /*reuse_b=*/false);

    /* ---- Epilogue: multiply by ts[t,n] and store ----
       C/D layout: VGPR r, lane<16: (M=r, N=lane); lane>=16: (M=r+8, N=lane-16) */
    const int n = n0 + wave * 16 + l15;
    if (n < HD_D) {
#pragma unroll
        for (int r = 0; r < 8; ++r) {
            const int m = m0 + r + 8 * lhalf;
            const int t = m & (HD_T - 1);
            samples[(size_t)m * HD_D + n] = acc[r] * ts[(size_t)t * HD_D + n];
        }
    }
}

/* ---------------------------------------------------------------------------
 * Phase 3: n-gram bind (rolled along D) + bundle over 125 positions.
 * hv[b,d] = sum_j  prod_{i=0..3} samples[b, j+i, (d-(3-i)) mod D]
 * ------------------------------------------------------------------------- */
__global__ __launch_bounds__(256) void ngram_kernel(
    const float* __restrict__ samples,    /* [512][D] */
    float* __restrict__ hv)               /* [4][D]   */
{
    const int i = blockIdx.x * 256 + threadIdx.x;
    if (i >= HD_B * HD_D) return;
    const int b = i / HD_D;
    const int d = i % HD_D;

    int d0 = d - 3; if (d0 < 0) d0 += HD_D;
    int d1 = d - 2; if (d1 < 0) d1 += HD_D;
    int d2 = d - 1; if (d2 < 0) d2 += HD_D;

    const float* S = samples + (size_t)b * HD_T * HD_D;
    float acc = 0.0f;
    for (int j = 0; j < HD_TP; ++j) {
        const float* row = S + (size_t)j * HD_D;
        const float p = row[d0]
                      * row[HD_D + d1]
                      * row[2 * HD_D + d2]
                      * row[3 * HD_D + d];
        acc += p;
    }
    hv[i] = acc;
}

/* ---------------------------------------------------------------------------
 * Phase 4: enc = sign(hv);  out[b,k] = sum_d enc[b,d] * classify_w[k,d]
 * One block per (b,k); LDS tree reduction.
 * ------------------------------------------------------------------------- */
__global__ __launch_bounds__(256) void classify_kernel(
    const float* __restrict__ hv,         /* [4][D]  */
    const float* __restrict__ cls,        /* [10][D] */
    float* __restrict__ out)              /* [4][10] */
{
    __shared__ float red[256];
    const int blk = blockIdx.x;           /* 0..39 */
    const int b = blk / HD_CLASSES;
    const int k = blk % HD_CLASSES;
    const int tid = threadIdx.x;

    float acc = 0.0f;
    for (int d = tid; d < HD_D; d += 256) {
        const float s = (hv[(size_t)b * HD_D + d] > 0.0f) ? 1.0f : -1.0f;
        acc += s * cls[(size_t)k * HD_D + d];
    }
    red[tid] = acc;
    __syncthreads();
#pragma unroll
    for (int off = 128; off > 0; off >>= 1) {
        if (tid < off) red[tid] += red[tid + off];
        __syncthreads();
    }
    if (tid == 0) out[blk] = red[0];
}

/* --------------------------------------------------------------------------- */
extern "C" void kernel_launch(void* const* d_in, const int* in_sizes, int n_in,
                              void* d_out, int out_size, void* d_ws, size_t ws_size,
                              hipStream_t stream) {
    const float* x    = (const float*)d_in[0];   /* [4,128,32]  */
    const float* sigw = (const float*)d_in[1];   /* [21,10000]  */
    /* d_in[2] = channels_w: dead bind in the reference, unused */
    const float* tsw  = (const float*)d_in[3];   /* [128,10000] */
    const float* clsw = (const float*)d_in[4];   /* [10,10000]  */
    float* out = (float*)d_out;                  /* [4,10] */

    /* workspace layout */
    char* ws = (char*)d_ws;
    const size_t OFF_COUNTS  = 0;                                   /* 512*32*2  = 32768   */
    const size_t OFF_SIG     = OFF_COUNTS + (size_t)HD_M * HD_KPAD * 2;
    const size_t OFF_SAMPLES = OFF_SIG + (size_t)HD_KPAD * HD_D * 2;   /* 640000 */
    const size_t OFF_HV      = OFF_SAMPLES + (size_t)HD_M * HD_D * 4;  /* 20.48 MB */

    _Float16* counts  = (_Float16*)(ws + OFF_COUNTS);
    _Float16* sig16   = (_Float16*)(ws + OFF_SIG);
    float*    samples = (float*)(ws + OFF_SAMPLES);
    float*    hv      = (float*)(ws + OFF_HV);

    /* Phase 1: histograms */
    counts_kernel<<<HD_M / 4, 128, 0, stream>>>(x, counts);

    /* Phase 2a: signals -> f16 panel */
    sigcvt_kernel<<<(HD_KPAD * HD_D + 255) / 256, 256, 0, stream>>>(sigw, sig16);

    /* Phase 2b: WMMA GEMM + ts epilogue */
    dim3 ggrid(HD_M / 16, (HD_D / 16 + 7) / 8);   /* (32, 79) */
    wmma_gemm_kernel<<<ggrid, 256, 0, stream>>>(counts, sig16, tsw, samples);

    /* Phase 3: n-gram bind + bundle */
    ngram_kernel<<<(HD_B * HD_D + 255) / 256, 256, 0, stream>>>(samples, hv);

    /* Phase 4: quantize + classify */
    classify_kernel<<<HD_B * HD_CLASSES, 256, 0, stream>>>(hv, clsw, out);
}
